// E2_EGNN_35682588295912
// MI455X (gfx1250) — matrix operand order repaired and checked by reading, not compiled, not run on hardware
//
#include <hip/hip_runtime.h>

#define NNODES 100000
#define NEDGES 1600000
#define EMB 128
#define HID 128

typedef __attribute__((ext_vector_type(16))) __bf16 v16bf;
typedef __attribute__((ext_vector_type(8)))  float  v8f;

union Frag16 { uint4 q[2]; v16bf v; };

__device__ __forceinline__ unsigned short f2bf(float f) {
    union { float f; unsigned u; } x; x.f = f;
    unsigned r = x.u + 0x7FFFu + ((x.u >> 16) & 1u);   // round-to-nearest-even
    return (unsigned short)(r >> 16);
}

// A fragment (16x32 bf16). ISA layout: lane L holds row M=L&15; element e maps to
// K = e + (e>=8 ? 8 : 0) + 8*(L>>4)  -> two contiguous 8-half (16B) groups.
__device__ __forceinline__ v16bf ldsA(const unsigned short* row, int kc, int g) {
    Frag16 u;
    u.q[0] = *(const uint4*)(row + kc * 32 + 8 * g);
    u.q[1] = *(const uint4*)(row + kc * 32 + 16 + 8 * g);
    return u.v;
}

// B fragment from pre-swizzled weights: per-lane 16 halves contiguous (32B).
__device__ __forceinline__ v16bf ldB(const unsigned short* Wp, int kc, int nNt, int nt, int lane) {
    Frag16 u;
    const uint4* p = (const uint4*)(Wp + (size_t)(((kc * nNt + nt) * 32 + lane) * 16));
    u.q[0] = p[0]; u.q[1] = p[1];
    return u.v;
}

// Pack W[K x N] (row-major f32) into bf16 B fragments.
// Fragment element e, lane -> K = kc*32 + (e&7) + 8*(lane>>4) + 16*(e>>3), N = nt*16 + (lane&15).
__global__ __launch_bounds__(256) void egnn_pack_b(const float* __restrict__ W,
                                                   int Kdim, int Ndim, int nNt, int total,
                                                   unsigned short* __restrict__ out) {
    int idx = blockIdx.x * 256 + threadIdx.x;
    if (idx >= total) return;
    int e    = idx & 15;
    int lane = (idx >> 4) & 31;
    int nt   = (idx >> 9) % nNt;
    int kc   = idx / (512 * nNt);
    int K = kc * 32 + (e & 7) + 8 * (lane >> 4) + 16 * (e >> 3);
    int N = nt * 16 + (lane & 15);
    float v = (K < Kdim && N < Ndim) ? W[(long)K * Ndim + N] : 0.0f;
    out[idx] = f2bf(v);
}

// d_out init: copy x into x_new region; zero h region (feature accumulator).
__global__ __launch_bounds__(256) void egnn_init_out(const float* __restrict__ x,
                                                     float* __restrict__ out) {
    long i = (long)blockIdx.x * 256 + threadIdx.x;
    if (i < (long)NNODES * 3)   out[i] = x[i];
    if (i < (long)NNODES * HID) out[(long)NNODES * 3 + i] = 0.0f;
}

// ---------------- Edge kernel: 4 waves/block, 32 edges per wave ----------------
// Each loaded B fragment feeds TWO 16x16x32 WMMAs (rows 0-15 and 16-31).
__global__ __launch_bounds__(128) void egnn_edge_kernel(
    const float* __restrict__ x, const float* __restrict__ h,
    const int* __restrict__ n1, const int* __restrict__ n2,
    const unsigned short* __restrict__ wMsg1, const float* __restrict__ bMsg1,
    const unsigned short* __restrict__ wMsg2, const float* __restrict__ bMsg2,
    const unsigned short* __restrict__ wCrd1, const float* __restrict__ bCrd1,
    const unsigned short* __restrict__ wCw2,  const float* __restrict__ bCw2,
    float* __restrict__ xOut, float* __restrict__ featAcc)
{
    __shared__ __align__(16) unsigned short sIn[4][32][288];   // msg_in (257 padded to 288)
    __shared__ __align__(16) unsigned short sMid[4][32][128];  // inter-layer bf16 staging
    __shared__ float sNd[4][32][3];
    __shared__ float sScale[4][32];
    __shared__ int   sN1[4][32];

    const int tid = threadIdx.x;
    const int w = tid >> 5;
    const int L = tid & 31;
    const int g = L >> 4;
    const int n = L & 15;
    const long ebase = (long)blockIdx.x * 128 + (long)w * 32;

    // ---- stage [h[n1] | h[n2] | radial | 0] as bf16 (wave-private, no barriers) ----
    for (int m = 0; m < 32; ++m) {
        long e  = ebase + m;
        long ec = (e < NEDGES) ? e : 0;
        int na = n1[ec], nb = n2[ec];
        float4 va = ((const float4*)(h + (long)na * EMB))[L];
        float4 vb = ((const float4*)(h + (long)nb * EMB))[L];
        unsigned short* row = sIn[w][m];
        row[L*4+0] = f2bf(va.x); row[L*4+1] = f2bf(va.y);
        row[L*4+2] = f2bf(va.z); row[L*4+3] = f2bf(va.w);
        row[128+L*4+0] = f2bf(vb.x); row[128+L*4+1] = f2bf(vb.y);
        row[128+L*4+2] = f2bf(vb.z); row[128+L*4+3] = f2bf(vb.w);
        for (int c = 256 + L; c < 288; c += 32) row[c] = 0;
        if (L == 0) sN1[w][m] = na;
    }
    {   // radial + normalized coord diff, edge m = L (all 32 lanes)
        long e  = ebase + L;
        long ec = (e < NEDGES) ? e : 0;
        int na = n1[ec], nb = n2[ec];
        float dx = x[(long)na*3+0] - x[(long)nb*3+0];
        float dy = x[(long)na*3+1] - x[(long)nb*3+1];
        float dz = x[(long)na*3+2] - x[(long)nb*3+2];
        float rad = dx*dx + dy*dy + dz*dz;
        float inv = rsqrtf(rad + 1e-8f);
        sNd[w][L][0] = dx*inv; sNd[w][L][1] = dy*inv; sNd[w][L][2] = dz*inv;
        sIn[w][L][256] = f2bf(rad);
    }

    const unsigned short* rowA0 = sIn[w][n];        // A row for tile 0 (edges 0-15)
    const unsigned short* rowA1 = sIn[w][16 + n];   // A row for tile 1 (edges 16-31)
    const unsigned short* rowM0 = sMid[w][n];
    const unsigned short* rowM1 = sMid[w][16 + n];

    // ---- msg MLP layer 1: [32x257]@[257x128]+b, relu -> sMid ----
    for (int nt = 0; nt < 8; ++nt) {
        float bias = bMsg1[nt*16 + n];
        v8f c0, c1;
#pragma unroll
        for (int r = 0; r < 8; ++r) { c0[r] = bias; c1[r] = bias; }
#pragma unroll
        for (int kc = 0; kc < 9; ++kc) {
            v16bf a0 = ldsA(rowA0, kc, g);
            v16bf a1 = ldsA(rowA1, kc, g);
            v16bf b  = ldB(wMsg1, kc, 8, nt, L);
            c0 = __builtin_amdgcn_wmma_f32_16x16x32_bf16(false, a0, false, b, (short)0, c0, false, false);
            c1 = __builtin_amdgcn_wmma_f32_16x16x32_bf16(false, a1, false, b, (short)0, c1, false, false);
        }
#pragma unroll
        for (int r = 0; r < 8; ++r) {
            sMid[w][r + 8*g][nt*16 + n]      = f2bf(fmaxf(c0[r], 0.0f));
            sMid[w][16 + r + 8*g][nt*16 + n] = f2bf(fmaxf(c1[r], 0.0f));
        }
    }

    // ---- msg MLP layer 2: +b, relu -> msgs; scatter to featAcc; restage bf16 ----
    v16bf aM0[4], aM1[4];
#pragma unroll
    for (int kc = 0; kc < 4; ++kc) { aM0[kc] = ldsA(rowM0, kc, g); aM1[kc] = ldsA(rowM1, kc, g); }
    for (int nt = 0; nt < 8; ++nt) {
        float bias = bMsg2[nt*16 + n];
        v8f c0, c1;
#pragma unroll
        for (int r = 0; r < 8; ++r) { c0[r] = bias; c1[r] = bias; }
#pragma unroll
        for (int kc = 0; kc < 4; ++kc) {
            v16bf b = ldB(wMsg2, kc, 8, nt, L);
            c0 = __builtin_amdgcn_wmma_f32_16x16x32_bf16(false, aM0[kc], false, b, (short)0, c0, false, false);
            c1 = __builtin_amdgcn_wmma_f32_16x16x32_bf16(false, aM1[kc], false, b, (short)0, c1, false, false);
        }
#pragma unroll
        for (int r = 0; r < 8; ++r) {
            float v0 = fmaxf(c0[r], 0.0f);
            float v1 = fmaxf(c1[r], 0.0f);
            int M0 = r + 8*g, M1 = 16 + r + 8*g;
            if (ebase + M0 < NEDGES)
                atomicAdd(&featAcc[(long)sN1[w][M0] * HID + nt*16 + n], v0);
            if (ebase + M1 < NEDGES)
                atomicAdd(&featAcc[(long)sN1[w][M1] * HID + nt*16 + n], v1);
            sMid[w][M0][nt*16 + n] = f2bf(v0);
            sMid[w][M1][nt*16 + n] = f2bf(v1);
        }
    }

    // ---- coord MLP layer 1: relu -> sMid ----
    v16bf aC0[4], aC1[4];
#pragma unroll
    for (int kc = 0; kc < 4; ++kc) { aC0[kc] = ldsA(rowM0, kc, g); aC1[kc] = ldsA(rowM1, kc, g); }
    for (int nt = 0; nt < 8; ++nt) {
        float bias = bCrd1[nt*16 + n];
        v8f c0, c1;
#pragma unroll
        for (int r = 0; r < 8; ++r) { c0[r] = bias; c1[r] = bias; }
#pragma unroll
        for (int kc = 0; kc < 4; ++kc) {
            v16bf b = ldB(wCrd1, kc, 8, nt, L);
            c0 = __builtin_amdgcn_wmma_f32_16x16x32_bf16(false, aC0[kc], false, b, (short)0, c0, false, false);
            c1 = __builtin_amdgcn_wmma_f32_16x16x32_bf16(false, aC1[kc], false, b, (short)0, c1, false, false);
        }
#pragma unroll
        for (int r = 0; r < 8; ++r) {
            sMid[w][r + 8*g][nt*16 + n]      = f2bf(fmaxf(c0[r], 0.0f));
            sMid[w][16 + r + 8*g][nt*16 + n] = f2bf(fmaxf(c1[r], 0.0f));
        }
    }

    // ---- coord MLP layer 2: [32x128]@[128x1] via col-0-padded tile, no relu ----
    v16bf aS0[4], aS1[4];
#pragma unroll
    for (int kc = 0; kc < 4; ++kc) { aS0[kc] = ldsA(rowM0, kc, g); aS1[kc] = ldsA(rowM1, kc, g); }
    {
        float b2 = bCw2[0];
        v8f c0, c1;
#pragma unroll
        for (int r = 0; r < 8; ++r) { float v = (n == 0) ? b2 : 0.0f; c0[r] = v; c1[r] = v; }
#pragma unroll
        for (int kc = 0; kc < 4; ++kc) {
            v16bf b = ldB(wCw2, kc, 1, 0, L);
            c0 = __builtin_amdgcn_wmma_f32_16x16x32_bf16(false, aS0[kc], false, b, (short)0, c0, false, false);
            c1 = __builtin_amdgcn_wmma_f32_16x16x32_bf16(false, aS1[kc], false, b, (short)0, c1, false, false);
        }
        if (n == 0) {
#pragma unroll
            for (int r = 0; r < 8; ++r) {
                sScale[w][r + 8*g]      = c0[r];
                sScale[w][16 + r + 8*g] = c1[r];
            }
        }
    }

    // ---- coord scatter: x_new += norm_coord_diff * scale (edge m = L) ----
    {
        long e = ebase + L;
        if (e < NEDGES) {
            float s = sScale[w][L];
            int node = sN1[w][L];
            atomicAdd(&xOut[(long)node*3+0], sNd[w][L][0] * s);
            atomicAdd(&xOut[(long)node*3+1], sNd[w][L][1] * s);
            atomicAdd(&xOut[(long)node*3+2], sNd[w][L][2] * s);
        }
    }
}

// ---------------- Node kernel: 4 waves/block, 32 nodes per wave ----------------
__global__ __launch_bounds__(128) void egnn_node_kernel(
    const float* __restrict__ h,
    const unsigned short* __restrict__ wNod1, const float* __restrict__ bNod1,
    const unsigned short* __restrict__ wNod2, const float* __restrict__ bNod2,
    float* __restrict__ hOut /* holds featAcc on entry, h_new on exit */)
{
    __shared__ __align__(16) unsigned short sIn[4][32][256];   // [feat | h] bf16
    __shared__ __align__(16) unsigned short sHid[4][32][128];

    const int tid = threadIdx.x;
    const int w = tid >> 5;
    const int L = tid & 31;
    const int g = L >> 4;
    const int n = L & 15;
    const long nbase = (long)blockIdx.x * 128 + (long)w * 32;

    for (int m = 0; m < 32; ++m) {
        long node = nbase + m;
        long nc = (node < NNODES) ? node : (NNODES - 1);
        float4 fv = ((const float4*)(hOut + nc * HID))[L];   // feature_updates
        float4 hv = ((const float4*)(h    + nc * EMB))[L];
        unsigned short* row = sIn[w][m];
        row[L*4+0] = f2bf(fv.x); row[L*4+1] = f2bf(fv.y);
        row[L*4+2] = f2bf(fv.z); row[L*4+3] = f2bf(fv.w);
        row[128+L*4+0] = f2bf(hv.x); row[128+L*4+1] = f2bf(hv.y);
        row[128+L*4+2] = f2bf(hv.z); row[128+L*4+3] = f2bf(hv.w);
    }

    const unsigned short* rowA0 = sIn[w][n];
    const unsigned short* rowA1 = sIn[w][16 + n];
    const unsigned short* rowM0 = sHid[w][n];
    const unsigned short* rowM1 = sHid[w][16 + n];

    // layer 1: [32x256]@[256x128]+b, relu
    for (int nt = 0; nt < 8; ++nt) {
        float bias = bNod1[nt*16 + n];
        v8f c0, c1;
#pragma unroll
        for (int r = 0; r < 8; ++r) { c0[r] = bias; c1[r] = bias; }
#pragma unroll
        for (int kc = 0; kc < 8; ++kc) {
            v16bf a0 = ldsA(rowA0, kc, g);
            v16bf a1 = ldsA(rowA1, kc, g);
            v16bf b  = ldB(wNod1, kc, 8, nt, L);
            c0 = __builtin_amdgcn_wmma_f32_16x16x32_bf16(false, a0, false, b, (short)0, c0, false, false);
            c1 = __builtin_amdgcn_wmma_f32_16x16x32_bf16(false, a1, false, b, (short)0, c1, false, false);
        }
#pragma unroll
        for (int r = 0; r < 8; ++r) {
            sHid[w][r + 8*g][nt*16 + n]      = f2bf(fmaxf(c0[r], 0.0f));
            sHid[w][16 + r + 8*g][nt*16 + n] = f2bf(fmaxf(c1[r], 0.0f));
        }
    }

    // layer 2: no relu; h_new = h + out (overwrites featAcc rows owned by this wave)
    v16bf aN0[4], aN1[4];
#pragma unroll
    for (int kc = 0; kc < 4; ++kc) { aN0[kc] = ldsA(rowM0, kc, g); aN1[kc] = ldsA(rowM1, kc, g); }
    for (int nt = 0; nt < 8; ++nt) {
        float bias = bNod2[nt*16 + n];
        v8f c0, c1;
#pragma unroll
        for (int r = 0; r < 8; ++r) { c0[r] = bias; c1[r] = bias; }
#pragma unroll
        for (int kc = 0; kc < 4; ++kc) {
            v16bf b = ldB(wNod2, kc, 8, nt, L);
            c0 = __builtin_amdgcn_wmma_f32_16x16x32_bf16(false, aN0[kc], false, b, (short)0, c0, false, false);
            c1 = __builtin_amdgcn_wmma_f32_16x16x32_bf16(false, aN1[kc], false, b, (short)0, c1, false, false);
        }
#pragma unroll
        for (int r = 0; r < 8; ++r) {
            long node0 = nbase + r + 8*g;
            long node1 = nbase + 16 + r + 8*g;
            if (node0 < NNODES) {
                long o = node0 * EMB + nt*16 + n;
                hOut[o] = h[o] + c0[r];
            }
            if (node1 < NNODES) {
                long o = node1 * EMB + nt*16 + n;
                hOut[o] = h[o] + c1[r];
            }
        }
    }
}

extern "C" void kernel_launch(void* const* d_in, const int* in_sizes, int n_in,
                              void* d_out, int out_size, void* d_ws, size_t ws_size,
                              hipStream_t stream)
{
    (void)in_sizes; (void)n_in; (void)out_size; (void)ws_size;

    const float* x        = (const float*)d_in[0];
    const float* h        = (const float*)d_in[1];
    const int*   ei       = (const int*)d_in[2];
    const float* msg_W1   = (const float*)d_in[3];
    const float* msg_b1   = (const float*)d_in[4];
    const float* msg_W2   = (const float*)d_in[5];
    const float* msg_b2   = (const float*)d_in[6];
    const float* coord_W1 = (const float*)d_in[7];
    const float* coord_b1 = (const float*)d_in[8];
    const float* coord_W2 = (const float*)d_in[9];
    const float* coord_b2 = (const float*)d_in[10];
    const float* node_W1  = (const float*)d_in[11];
    const float* node_b1  = (const float*)d_in[12];
    const float* node_W2  = (const float*)d_in[13];
    const float* node_b2  = (const float*)d_in[14];

    const int* n1 = ei;
    const int* n2 = ei + NEDGES;

    unsigned short* wsU   = (unsigned short*)d_ws;
    unsigned short* wMsg1 = wsU;            // 9*8*512 = 36864 halves
    unsigned short* wMsg2 = wsU + 36864;    // 4*8*512 = 16384
    unsigned short* wCrd1 = wsU + 53248;    // 16384
    unsigned short* wCw2  = wsU + 69632;    // 4*1*512 = 2048
    unsigned short* wNod1 = wsU + 71680;    // 8*8*512 = 32768
    unsigned short* wNod2 = wsU + 104448;   // 16384  (total 120832 halves = 242KB)

    float* xOut = (float*)d_out;
    float* hOut = xOut + (long)NNODES * 3;  // doubles as feature accumulator

    egnn_init_out<<<50000, 256, 0, stream>>>(x, xOut);

    egnn_pack_b<<<144, 256, 0, stream>>>(msg_W1,   257, 128, 8, 36864, wMsg1);
    egnn_pack_b<<<64,  256, 0, stream>>>(msg_W2,   128, 128, 8, 16384, wMsg2);
    egnn_pack_b<<<64,  256, 0, stream>>>(coord_W1, 128, 128, 8, 16384, wCrd1);
    egnn_pack_b<<<8,   256, 0, stream>>>(coord_W2, 128,   1, 1,  2048, wCw2);
    egnn_pack_b<<<128, 256, 0, stream>>>(node_W1,  256, 128, 8, 32768, wNod1);
    egnn_pack_b<<<64,  256, 0, stream>>>(node_W2,  128, 128, 8, 16384, wNod2);

    egnn_edge_kernel<<<NEDGES / 128, 128, 0, stream>>>(
        x, h, n1, n2,
        wMsg1, msg_b1, wMsg2, msg_b2,
        wCrd1, coord_b1, wCw2, coord_b2,
        xOut, hOut);

    egnn_node_kernel<<<(NNODES + 127) / 128, 128, 0, stream>>>(
        h, wNod1, node_b1, wNod2, node_b2, hOut);
}
